// MoE_50225347559548
// MI455X (gfx1250) — matrix operand order repaired and checked by reading, not compiled, not run on hardware
//
#include <hip/hip_runtime.h>
#include <hip/hip_bf16.h>
#include <stdint.h>

// ---------------------------------------------------------------------------
// MoE (top-2 of 8 experts), D=1024, F=4096, T=8192.
// Compute-bound => bf16 WMMA (v_wmma_f32_16x16x32_bf16) with fp32 accumulate.
// GEMM2 A-tile (bf16 h) staged via GLOBAL_LOAD_ASYNC_TO_LDS_B64 (ASYNCcnt).
// fp32->bf16 conversions use native bf16 casts (hardware v_cvt, RNE).
// ---------------------------------------------------------------------------

typedef __attribute__((ext_vector_type(16))) __bf16 v16bf;
typedef __attribute__((ext_vector_type(8)))  __bf16 v8bf;
typedef __attribute__((ext_vector_type(2)))  __bf16 v2bf;
typedef __attribute__((ext_vector_type(8)))  float  v8f;

// native fp32 -> bf16 (RNE), bits in a uint16
__device__ __forceinline__ uint16_t f2bf_bits(float f) {
    union { __bf16 b; uint16_t u; } c; c.b = (__bf16)f;
    return c.u;
}
// pack two fp32 -> bf16x2 in one uint32 (backend emits packed cvt)
__device__ __forceinline__ uint32_t f2bf_pk(float lo, float hi) {
    v2bf v; v[0] = (__bf16)lo; v[1] = (__bf16)hi;
    union { v2bf v; uint32_t u; } c; c.v = v;
    return c.u;
}

__device__ __forceinline__ float gelu_f(float v) {
    const float k0 = 0.7978845608028654f;          // sqrt(2/pi)
    const float k1 = 0.044715f;
    float t = tanhf(k0 * (v + k1 * v * v * v));
    return 0.5f * v * (1.0f + t);
}

// Build a 16-element bf16 fragment from two 16-byte LDS chunks (2x ds_load_b128)
__device__ __forceinline__ v16bf load_frag(const uint16_t* p0, const uint16_t* p1) {
    v8bf lo = *(const v8bf*)p0;
    v8bf hi = *(const v8bf*)p1;
    return __builtin_shufflevector(lo, hi, 0,1,2,3,4,5,6,7,8,9,10,11,12,13,14,15);
}

// CDNA5 async global->LDS copy (8 bytes/lane), SADDR form; tracked by ASYNCcnt
__device__ __forceinline__ void async_copy_b64(uint32_t lds_off, const void* sbase,
                                               uint32_t voff) {
#if defined(__HIP_DEVICE_COMPILE__)
    asm volatile("global_load_async_to_lds_b64 %0, %1, %2"
                 :: "v"(lds_off), "v"(voff), "s"(sbase) : "memory");
#endif
}
__device__ __forceinline__ void wait_async0() {
#if defined(__HIP_DEVICE_COMPILE__)
    asm volatile("s_wait_asynccnt 0" ::: "memory");
#endif
}

// ---------------------------------------------------------------------------
// Kernel 0: zero d_out
// ---------------------------------------------------------------------------
__global__ void __launch_bounds__(256) moe_zero_kernel(float* __restrict__ p, size_t n4) {
    size_t i = (size_t)blockIdx.x * blockDim.x + threadIdx.x;
    if (i < n4) {
        float4 z; z.x = 0.f; z.y = 0.f; z.z = 0.f; z.w = 0.f;
        ((float4*)p)[i] = z;
    }
}

// ---------------------------------------------------------------------------
// Kernel 1: router -> dense top-2 gates [T, E]; one wave per token
// ---------------------------------------------------------------------------
__global__ void __launch_bounds__(256) moe_router_kernel(
    const float* __restrict__ x, const float* __restrict__ rw,
    float* __restrict__ gates, int T, int D, int E)
{
    int token = blockIdx.x * 8 + (threadIdx.x >> 5);
    int lane  = threadIdx.x & 31;
    if (token >= T) return;

    float acc[8];
#pragma unroll
    for (int e = 0; e < 8; ++e) acc[e] = 0.f;

    const float* xr = x + (size_t)token * D;
    for (int d = lane; d < D; d += 32) {
        float xv = xr[d];
        const float4 w0 = *(const float4*)(rw + (size_t)d * 8);
        const float4 w1 = *(const float4*)(rw + (size_t)d * 8 + 4);
        acc[0] += xv * w0.x; acc[1] += xv * w0.y;
        acc[2] += xv * w0.z; acc[3] += xv * w0.w;
        acc[4] += xv * w1.x; acc[5] += xv * w1.y;
        acc[6] += xv * w1.z; acc[7] += xv * w1.w;
    }
#pragma unroll
    for (int e = 0; e < 8; ++e)
#pragma unroll
        for (int off = 16; off > 0; off >>= 1)
            acc[e] += __shfl_xor(acc[e], off, 32);

    if (lane == 0) {
        float m = acc[0];
#pragma unroll
        for (int e = 1; e < 8; ++e) m = fmaxf(m, acc[e]);
        float p[8];
#pragma unroll
        for (int e = 0; e < 8; ++e) p[e] = __expf(acc[e] - m);
        int i1 = 0;                                   // first index wins ties
#pragma unroll
        for (int e = 1; e < 8; ++e) if (p[e] > p[i1]) i1 = e;
        int i2 = (i1 == 0) ? 1 : 0;
#pragma unroll
        for (int e = 0; e < 8; ++e) if (e != i1 && p[e] > p[i2]) i2 = e;
        float inv = 1.0f / (p[i1] + p[i2]);
#pragma unroll
        for (int e = 0; e < 8; ++e) {
            float g = (e == i1) ? p[i1] * inv : ((e == i2) ? p[i2] * inv : 0.f);
            gates[(size_t)token * E + e] = g;
        }
    }
}

// ---------------------------------------------------------------------------
// WMMA GEMM: C[M,N] = op(A[M,K] @ B[K,N])
//   A: fp32 (x, converted) or bf16-bits (h, async-DMA'd straight to LDS)
//   B: fp32 weights, converted + transposed into LDS on the fly
//   DO_GELU: C = bf16(gelu(acc)) -> h ;  DO_GATE: C(f32) += gate * acc -> out
// Block: 256 threads = 8 waves; tile 128x128; BK=32; wave = 4x2 16x16 subtiles
// ---------------------------------------------------------------------------
#define BM 128
#define BN 128
#define BK 32
#define LDA 40   // 32 + 8 bf16 pad: 16B-aligned rows, conflict-free frag reads
#define LDB 40

template<bool A_IS_BF16, bool DO_GELU, bool DO_GATE>
__global__ void __launch_bounds__(256) moe_gemm_wmma(
    const void* __restrict__ Ag_, const float* __restrict__ Bg,
    void* __restrict__ Cg_, const float* __restrict__ gates,
    int M, int N, int K, int expert, int E)
{
    __shared__ __align__(16) uint16_t As[BM * LDA];
    __shared__ __align__(16) uint16_t Bs[BN * LDB];

    const int tid   = threadIdx.x;
    const int mBase = blockIdx.y * BM;
    const int nBase = blockIdx.x * BN;
    const int wave  = tid >> 5;
    const int lane  = tid & 31;
    const int waveM = wave >> 2;          // 0..1  (64 rows each)
    const int waveN = wave & 3;           // 0..3  (32 cols each)
    const int lr    = lane & 15;
    const int lh    = lane >> 4;

    v8f acc[4][2];
#pragma unroll
    for (int i = 0; i < 4; ++i)
#pragma unroll
        for (int j = 0; j < 2; ++j)
#pragma unroll
            for (int q = 0; q < 8; ++q) acc[i][j][q] = 0.f;

    const int arow0 = tid >> 3;           // 0..31 (A: 8 thr/row x 4 elem)
    const int acol  = (tid & 7) << 2;     // 0..28
    const int bn0   = (tid & 31) << 2;    // 0..124 (B: 32 thr/row x 4 elem)
    const int bk0   = tid >> 5;           // 0..7

    // hoisted per-thread global pointers (advance by BK / BK*N per k-tile)
    const float* aF   = (const float*)Ag_ + (size_t)(mBase + arow0) * K + acol;
    const float* bF   = Bg + (size_t)bk0 * N + nBase + bn0;
    uint32_t     aOff = (uint32_t)(((size_t)(mBase + arow0) * K + acol) * 2u);
    const uint32_t aRow32B = (uint32_t)(32u * (uint32_t)K * 2u);
    const size_t   aRow32F = (size_t)32 * K;
    const size_t   bRow8   = (size_t)8 * N;
    // LDS byte offset of this thread's A staging slot (low 32 bits of generic ptr)
    const uint32_t ldsA0   = (uint32_t)(uintptr_t)(&As[arow0 * LDA + acol]);

    for (int kt = 0; kt < K; kt += BK) {
        // ---- stage A tile [128][32] bf16 into LDS ----
        if (A_IS_BF16) {
            // pure copy: CDNA5 async DMA, no VGPR round-trip, ASYNCcnt-tracked
#pragma unroll
            for (int p = 0; p < 4; ++p)
                async_copy_b64(ldsA0 + (uint32_t)(p * 32 * LDA * 2), Ag_,
                               aOff + (uint32_t)p * aRow32B);
        } else {
#pragma unroll
            for (int p = 0; p < 4; ++p) {
                int r = arow0 + p * 32;
                const float4 v = *(const float4*)(aF + p * aRow32F);
                uint2 o;
                o.x = f2bf_pk(v.x, v.y);
                o.y = f2bf_pk(v.z, v.w);
                *(uint2*)(&As[r * LDA + acol]) = o;
            }
        }
        // ---- stage B tile [32][128] fp32 -> LDS transposed [n][k] bf16 ----
#pragma unroll
        for (int p = 0; p < 4; ++p) {
            int k = bk0 + p * 8;
            const float4 v = *(const float4*)(bF + p * bRow8);
            Bs[(bn0 + 0) * LDB + k] = f2bf_bits(v.x);
            Bs[(bn0 + 1) * LDB + k] = f2bf_bits(v.y);
            Bs[(bn0 + 2) * LDB + k] = f2bf_bits(v.z);
            Bs[(bn0 + 3) * LDB + k] = f2bf_bits(v.w);
        }
        // hint next tiles into cache (global_prefetch_b8)
        if (kt + BK < K) {
            if (!A_IS_BF16) __builtin_prefetch(aF + BK, 0, 3);
            __builtin_prefetch(bF + (size_t)BK * N, 0, 3);
        }
        if (A_IS_BF16) wait_async0();
        __syncthreads();

        // ---- fragments per ISA 7.12.2 16-bit layouts ----
        v16bf afr[4];
#pragma unroll
        for (int i = 0; i < 4; ++i) {
            const uint16_t* base = &As[(waveM * 64 + i * 16 + lr) * LDA];
            afr[i] = load_frag(base + 8 * lh, base + 16 + 8 * lh); // K=8h.., 16+8h..
        }
        v16bf bfr[2];
#pragma unroll
        for (int j = 0; j < 2; ++j) {
            const uint16_t* base = &Bs[(waveN * 32 + j * 16 + lr) * LDB];
            bfr[j] = load_frag(base + 8 * lh, base + 16 + 8 * lh); // K=8h.., 16+8h..
        }
#pragma unroll
        for (int i = 0; i < 4; ++i)
#pragma unroll
            for (int j = 0; j < 2; ++j)
                acc[i][j] = __builtin_amdgcn_wmma_f32_16x16x32_bf16(
                    false, afr[i], false, bfr[j], (short)0, acc[i][j], false, false);
        __syncthreads();

        aF   += BK;
        aOff += (uint32_t)(BK * 2);
        bF   += (size_t)BK * N;
    }

    // ---- epilogue ----
    if (DO_GELU) {
        uint16_t* C16 = (uint16_t*)Cg_;
#pragma unroll
        for (int i = 0; i < 4; ++i) {
#pragma unroll
            for (int j = 0; j < 2; ++j) {
                int col = nBase + waveN * 32 + j * 16 + lr;
#pragma unroll
                for (int q = 0; q < 8; ++q) {
                    int row = mBase + waveM * 64 + i * 16 + lh * 8 + q;
                    C16[(size_t)row * N + col] = f2bf_bits(gelu_f(acc[i][j][q]));
                }
            }
        }
    }
    if (DO_GATE) {
        float* Cf = (float*)Cg_;
#pragma unroll
        for (int i = 0; i < 4; ++i) {
            float gv[8];
#pragma unroll
            for (int q = 0; q < 8; ++q) {
                int row = mBase + waveM * 64 + i * 16 + lh * 8 + q;
                gv[q] = gates[(size_t)row * E + expert];
            }
#pragma unroll
            for (int j = 0; j < 2; ++j) {
                int col = nBase + waveN * 32 + j * 16 + lr;
#pragma unroll
                for (int q = 0; q < 8; ++q) {
                    int row = mBase + waveM * 64 + i * 16 + lh * 8 + q;
                    size_t idx = (size_t)row * N + col;
                    Cf[idx] += gv[q] * acc[i][j][q];
                }
            }
        }
    }
}

// ---------------------------------------------------------------------------
extern "C" void kernel_launch(void* const* d_in, const int* in_sizes, int n_in,
                              void* d_out, int out_size, void* d_ws, size_t ws_size,
                              hipStream_t stream) {
    const float* x        = (const float*)d_in[0];  // [T, D]
    const float* router_w = (const float*)d_in[1];  // [D, E]
    const float* w1       = (const float*)d_in[2];  // [E, D, F]
    const float* w2       = (const float*)d_in[3];  // [E, F, D]
    float* out            = (float*)d_out;          // [T, D]

    const int E = 8;
    const int D = in_sizes[1] / E;                  // 1024
    const int T = in_sizes[0] / D;                  // 8192
    const int F = in_sizes[2] / (E * D);            // 4096

    // workspace carve-up
    float*    gates = (float*)d_ws;                                   // T*E f32
    size_t    goff  = (((size_t)T * E * sizeof(float)) + 255) & ~(size_t)255;
    uint16_t* h     = (uint16_t*)((char*)d_ws + goff);                // T*F bf16

    // 0) zero output (accumulated across experts)
    {
        size_t n4 = (size_t)T * D / 4;
        moe_zero_kernel<<<(unsigned)((n4 + 255) / 256), 256, 0, stream>>>(out, n4);
    }
    // 1) router -> gates
    moe_router_kernel<<<T / 8, 256, 0, stream>>>(x, router_w, gates, T, D, E);

    // 2) expert FFNs (stream-ordered; GEMM2[e] consumes h written by GEMM1[e])
    for (int e = 0; e < E; ++e) {
        dim3 g1(F / BN, T / BM);   // h = gelu(x @ w1[e])          -> bf16 h
        moe_gemm_wmma<false, true, false><<<g1, 256, 0, stream>>>(
            (const void*)x, w1 + (size_t)e * D * F, (void*)h, nullptr,
            T, F, D, e, E);
        dim3 g2(D / BN, T / BM);   // out += gate[:,e] * (h @ w2[e])
        moe_gemm_wmma<true, false, true><<<g2, 256, 0, stream>>>(
            (const void*)h, w2 + (size_t)e * F * D, (void*)out, gates,
            T, D, F, e, E);
    }
}
